// SuperpixelGCN_46866683134517
// MI455X (gfx1250) — matrix operand order repaired and checked by reading, not compiled, not run on hardware
//
#include <hip/hip_runtime.h>
#include <hip/hip_bf16.h>

#define N_NODES  100000
#define N_EDGES  3200000
#define N_GRAPHS 512
#define F_IN     128
#define F_HID    64
#define F_OUT    10
#define F_CAT    192   // 3 * F_HID

typedef __attribute__((ext_vector_type(16))) _Float16 v16h;
typedef __attribute__((ext_vector_type(8)))  _Float16 v8h;
typedef __attribute__((ext_vector_type(8)))  float    v8f;

// ---------------- utility ----------------
__global__ void k_zero(float* __restrict__ p, int n) {
    int i = blockIdx.x * blockDim.x + threadIdx.x;
    if (i < n) p[i] = 0.f;
}

// deg[col[e]] += 1
__global__ void k_deg(const int* __restrict__ col, float* __restrict__ deg) {
    int e = blockIdx.x * blockDim.x + threadIdx.x;
    if (e < N_EDGES) atomicAdd(&deg[col[e]], 1.0f);
}

// in-place: d = (deg>0) ? rsqrt(max(deg,1)) : 0
__global__ void k_dinv(float* __restrict__ d) {
    int i = blockIdx.x * blockDim.x + threadIdx.x;
    if (i < N_NODES) {
        float v = d[i];
        d[i] = (v > 0.f) ? rsqrtf(fmaxf(v, 1.f)) : 0.f;
    }
}

__global__ void k_norm(const int* __restrict__ row, const int* __restrict__ col,
                       const float* __restrict__ dinv, float* __restrict__ norm) {
    int e = blockIdx.x * blockDim.x + threadIdx.x;
    if (e < N_EDGES) norm[e] = dinv[row[e]] * dinv[col[e]];
}

__global__ void k_f32tof16(const float* __restrict__ src, _Float16* __restrict__ dst, int n) {
    int i = blockIdx.x * blockDim.x + threadIdx.x;
    if (i < n) dst[i] = (_Float16)src[i];
}

// Pack W [K x 64] f32 into WMMA B-operand lane order, f16.
// Layout: Wp[ ((blk*4 + nt)*32 + lane)*16 + i ] = W[(blk*32 + (i>>3)*16 + (lane>>4)*8 + (i&7))*64 + nt*16 + (lane&15)]
__global__ void k_packW(const float* __restrict__ W, _Float16* __restrict__ Wp, int K) {
    int tid = blockIdx.x * blockDim.x + threadIdx.x;
    int total = (K >> 5) * 4 * 32 * 16;
    if (tid >= total) return;
    int i    = tid & 15;
    int lane = (tid >> 4) & 31;
    int nt   = (tid >> 9) & 3;
    int blk  = tid >> 11;
    int n = nt * 16 + (lane & 15);
    int k = blk * 32 + ((i >> 3) * 16) + ((lane >> 4) * 8) + (i & 7);
    Wp[tid] = (_Float16)W[k * F_HID + n];
}

// ---------------- WMMA GEMM: H[M x 64] = A[M x K] (f16) @ Wp (packed f16), f32 accum ----
// One wave -> 16x64 tile. M = 100000 = 6250 tiles, 2 waves/block, 3125 blocks (EXEC always full).
__global__ void k_gemm(const _Float16* __restrict__ Ah, const _Float16* __restrict__ Wp,
                       float* __restrict__ Hout, int K) {
    const int lane = threadIdx.x & 31;
    const int wave = threadIdx.x >> 5;
    const int mt   = (blockIdx.x * 2 + wave) * 16;
    const int m    = mt + (lane & 15);
    const int base0 = (lane >> 4) * 8;   // K-offset of this lane's A halves

    v8f acc0 = {}, acc1 = {}, acc2 = {}, acc3 = {};
    const v16h* bp = (const v16h*)Wp;
    const int nblk = K >> 5;

    for (int blk = 0; blk < nblk; ++blk) {
        // A tile: lane m = L&15 ; a[0..7] = A[m, k0+base0 .. +7], a[8..15] = A[m, k0+16+base0 .. +7]
        const _Float16* ap = Ah + (size_t)m * K + blk * 32 + base0;
        v8h lo = *(const v8h*)(ap);
        v8h hi = *(const v8h*)(ap + 16);
        v16h a = __builtin_shufflevector(lo, hi, 0,1,2,3,4,5,6,7,8,9,10,11,12,13,14,15);

        v16h b0 = bp[(blk * 4 + 0) * 32 + lane];
        v16h b1 = bp[(blk * 4 + 1) * 32 + lane];
        v16h b2 = bp[(blk * 4 + 2) * 32 + lane];
        v16h b3 = bp[(blk * 4 + 3) * 32 + lane];

        acc0 = __builtin_amdgcn_wmma_f32_16x16x32_f16(false, a, false, b0, (short)0, acc0, false, false);
        acc1 = __builtin_amdgcn_wmma_f32_16x16x32_f16(false, a, false, b1, (short)0, acc1, false, false);
        acc2 = __builtin_amdgcn_wmma_f32_16x16x32_f16(false, a, false, b2, (short)0, acc2, false, false);
        acc3 = __builtin_amdgcn_wmma_f32_16x16x32_f16(false, a, false, b3, (short)0, acc3, false, false);
    }

    // C/D layout: VGPR j -> row (lane<16 ? j : 8+j), col = lane&15
    const int roff = (lane >> 4) * 8;
    const int c    = lane & 15;
#pragma unroll
    for (int j = 0; j < 8; ++j) {
        float* out = Hout + (size_t)(mt + roff + j) * F_HID + c;
        out[0]  = acc0[j];
        out[16] = acc1[j];
        out[32] = acc2[j];
        out[48] = acc3[j];
    }
}

// ---------------- message passing: Agg[col[e]] += norm[e] * H[row[e]] --------------------
// One wave per edge; lane handles 2 of 64 features (coalesced float2 gather, f32 atomics).
__global__ void k_scatter(const int* __restrict__ row, const int* __restrict__ col,
                          const float* __restrict__ norm, const float* __restrict__ H,
                          float* __restrict__ Agg) {
    const int lane   = threadIdx.x & 31;
    const int nwaves = (gridDim.x * blockDim.x) >> 5;
    int e = (blockIdx.x * blockDim.x + threadIdx.x) >> 5;
    for (; e < N_EDGES; e += nwaves) {
        int e2 = e + nwaves;
        if (e2 < N_EDGES)  // prefetch next source row (global_prefetch_b8)
            __builtin_prefetch(H + (size_t)row[e2] * F_HID + lane * 2, 0, 0);
        int   r = row[e];
        int   c = col[e];
        float w = norm[e];
        float2 v = *(const float2*)(H + (size_t)r * F_HID + lane * 2);
        float* dst = Agg + (size_t)c * F_HID + lane * 2;
        atomicAdd(dst,     v.x * w);
        atomicAdd(dst + 1, v.y * w);
    }
}

// y = relu(Agg + b); write f32 into xcat column block, and f16 copy for next layer's GEMM
__global__ void k_bias_relu(const float* __restrict__ Agg, const float* __restrict__ bias,
                            float* __restrict__ xcat_col, _Float16* __restrict__ Hh) {
    int i = blockIdx.x * blockDim.x + threadIdx.x;
    if (i >= N_NODES * F_HID) return;
    int node = i >> 6;
    int f    = i & 63;
    float y = fmaxf(Agg[i] + bias[f], 0.f);
    xcat_col[(size_t)node * F_CAT + f] = y;
    if (Hh) Hh[i] = (_Float16)y;
}

// mean-pool accumulation: one wave per node, 6 features per lane
__global__ void k_pool(const int* __restrict__ batch, const float* __restrict__ xcat,
                       float* __restrict__ sums, float* __restrict__ cnt) {
    int t    = blockIdx.x * blockDim.x + threadIdx.x;
    int node = t >> 5;
    int lane = t & 31;
    if (node >= N_NODES) return;
    int g = batch[node];
#pragma unroll
    for (int q = 0; q < 6; ++q) {
        int f = lane + q * 32;
        atomicAdd(&sums[(size_t)g * F_CAT + f], xcat[(size_t)node * F_CAT + f]);
    }
    if (lane == 0) atomicAdd(&cnt[g], 1.0f);
}

// pooled = sums/cnt ; logits = pooled @ Wf + bf ; softmax
__global__ void k_final(const float* __restrict__ sums, const float* __restrict__ cnt,
                        const float* __restrict__ Wf, const float* __restrict__ bf,
                        float* __restrict__ out) {
    int g = blockIdx.x * blockDim.x + threadIdx.x;
    if (g >= N_GRAPHS) return;
    float inv = 1.0f / fmaxf(cnt[g], 1.0f);
    float logits[F_OUT];
#pragma unroll
    for (int c = 0; c < F_OUT; ++c) logits[c] = bf[c];
    for (int f = 0; f < F_CAT; ++f) {
        float p = sums[(size_t)g * F_CAT + f] * inv;
#pragma unroll
        for (int c = 0; c < F_OUT; ++c)
            logits[c] = fmaf(p, Wf[f * F_OUT + c], logits[c]);
    }
    float mx = logits[0];
#pragma unroll
    for (int c = 1; c < F_OUT; ++c) mx = fmaxf(mx, logits[c]);
    float s = 0.f, e_[F_OUT];
#pragma unroll
    for (int c = 0; c < F_OUT; ++c) { e_[c] = __expf(logits[c] - mx); s += e_[c]; }
    float is = 1.0f / s;
#pragma unroll
    for (int c = 0; c < F_OUT; ++c) out[(size_t)g * F_OUT + c] = e_[c] * is;
}

extern "C" void kernel_launch(void* const* d_in, const int* in_sizes, int n_in,
                              void* d_out, int out_size, void* d_ws, size_t ws_size,
                              hipStream_t stream) {
    (void)in_sizes; (void)n_in; (void)out_size; (void)ws_size;
    const float* X0 = (const float*)d_in[0];
    const int*   ei = (const int*)d_in[1];
    const int*   bt = (const int*)d_in[2];
    const float* W1 = (const float*)d_in[3];  const float* b1 = (const float*)d_in[4];
    const float* W2 = (const float*)d_in[5];  const float* b2 = (const float*)d_in[6];
    const float* W3 = (const float*)d_in[7];  const float* b3 = (const float*)d_in[8];
    const float* Wf = (const float*)d_in[9];  const float* bf = (const float*)d_in[10];
    const int* row = ei;
    const int* col = ei + N_EDGES;

    // workspace carve-out (256B-aligned chunks)
    char* ws = (char*)d_ws;
    size_t off = 0;
    auto alloc = [&](size_t bytes) -> char* {
        char* p = ws + off;
        off = (off + bytes + 255) & ~(size_t)255;
        return p;
    };
    float*    dinv = (float*)   alloc((size_t)N_NODES * 4);
    float*    nrm  = (float*)   alloc((size_t)N_EDGES * 4);
    _Float16* Xh   = (_Float16*)alloc((size_t)N_NODES * F_IN * 2);  // reused as N x 64 f16 later
    _Float16* Wp1  = (_Float16*)alloc((size_t)(F_IN  >> 5) * 2048 * 2);
    _Float16* Wp2  = (_Float16*)alloc((size_t)(F_HID >> 5) * 2048 * 2);
    _Float16* Wp3  = (_Float16*)alloc((size_t)(F_HID >> 5) * 2048 * 2);
    float*    Hout = (float*)   alloc((size_t)N_NODES * F_HID * 4);
    float*    Agg  = (float*)   alloc((size_t)N_NODES * F_HID * 4);
    float*    xcat = (float*)   alloc((size_t)N_NODES * F_CAT * 4);
    float*    sums = (float*)   alloc((size_t)N_GRAPHS * F_CAT * 4);
    float*    cnt  = (float*)   alloc((size_t)N_GRAPHS * 4);

    const int TB = 256;
    const int gN   = (N_NODES + TB - 1) / TB;
    const int gE   = (N_EDGES + TB - 1) / TB;
    const int gNF  = (N_NODES * F_HID + TB - 1) / TB;
    const int gXin = (N_NODES * F_IN + TB - 1) / TB;

    // normalization
    k_zero<<<gN, TB, 0, stream>>>(dinv, N_NODES);
    k_deg <<<gE, TB, 0, stream>>>(col, dinv);
    k_dinv<<<gN, TB, 0, stream>>>(dinv);
    k_norm<<<gE, TB, 0, stream>>>(row, col, dinv, nrm);

    // precision staging
    k_f32tof16<<<gXin, TB, 0, stream>>>(X0, Xh, N_NODES * F_IN);
    k_packW<<<((F_IN  >> 5) * 2048 + TB - 1) / TB, TB, 0, stream>>>(W1, Wp1, F_IN);
    k_packW<<<((F_HID >> 5) * 2048 + TB - 1) / TB, TB, 0, stream>>>(W2, Wp2, F_HID);
    k_packW<<<((F_HID >> 5) * 2048 + TB - 1) / TB, TB, 0, stream>>>(W3, Wp3, F_HID);

    const int GEMM_BLOCKS = (N_NODES / 16) / 2;   // 3125, exact
    const int SCAT_BLOCKS = 4096;

    // layer 1
    k_gemm<<<GEMM_BLOCKS, 64, 0, stream>>>(Xh, Wp1, Hout, F_IN);
    k_zero<<<gNF, TB, 0, stream>>>(Agg, N_NODES * F_HID);
    k_scatter<<<SCAT_BLOCKS, TB, 0, stream>>>(row, col, nrm, Hout, Agg);
    k_bias_relu<<<gNF, TB, 0, stream>>>(Agg, b1, xcat + 0, Xh);

    // layer 2
    k_gemm<<<GEMM_BLOCKS, 64, 0, stream>>>(Xh, Wp2, Hout, F_HID);
    k_zero<<<gNF, TB, 0, stream>>>(Agg, N_NODES * F_HID);
    k_scatter<<<SCAT_BLOCKS, TB, 0, stream>>>(row, col, nrm, Hout, Agg);
    k_bias_relu<<<gNF, TB, 0, stream>>>(Agg, b2, xcat + 64, Xh);

    // layer 3
    k_gemm<<<GEMM_BLOCKS, 64, 0, stream>>>(Xh, Wp3, Hout, F_HID);
    k_zero<<<gNF, TB, 0, stream>>>(Agg, N_NODES * F_HID);
    k_scatter<<<SCAT_BLOCKS, TB, 0, stream>>>(row, col, nrm, Hout, Agg);
    k_bias_relu<<<gNF, TB, 0, stream>>>(Agg, b3, xcat + 128, (_Float16*)nullptr);

    // pooling + head
    k_zero<<<(N_GRAPHS * F_CAT + TB - 1) / TB, TB, 0, stream>>>(sums, N_GRAPHS * F_CAT);
    k_zero<<<(N_GRAPHS + TB - 1) / TB, TB, 0, stream>>>(cnt, N_GRAPHS);
    k_pool<<<(N_NODES * 32 + TB - 1) / TB, TB, 0, stream>>>(bt, xcat, sums, cnt);
    k_final<<<(N_GRAPHS + TB - 1) / TB, TB, 0, stream>>>(sums, cnt, Wf, bf, (float*)d_out);
}
